// Head_58317065945739
// MI455X (gfx1250) — matrix operand order repaired and checked by reading, not compiled, not run on hardware
//
#include <hip/hip_runtime.h>
#include <hip/hip_bf16.h>

typedef _Float16 h8   __attribute__((ext_vector_type(8)));
typedef _Float16 v16h __attribute__((ext_vector_type(16)));
typedef float    v8f  __attribute__((ext_vector_type(8)));
typedef float    f4   __attribute__((ext_vector_type(4)));

constexpr int NB = 4;
constexpr int NT = 4096;
constexpr int NC = 1024;
constexpr int NH = 64;

// ---------------------------------------------------------------------------
// Kernel 0: one-shot weight conversion f32 -> f16, tiled+transposed:
// WhT[w][kc 0..31][n 0..63][k 0..31]  (each [n][k] tile = 4KB, contiguous)
// ---------------------------------------------------------------------------
__global__ __launch_bounds__(256) void conv_weights(
    const float* __restrict__ Wq, const float* __restrict__ Wk,
    const float* __restrict__ Wv, _Float16* __restrict__ WhT)
{
  int base = (blockIdx.x * 256 + threadIdx.x) * 4;
#pragma unroll
  for (int i = 0; i < 4; ++i) {
    int dst = base + i;                      // 0 .. 3*32*64*32-1
    int w   = dst >> 16;
    int r   = dst & 65535;
    int kc  = r >> 11;
    int r2  = r & 2047;
    int n   = r2 >> 5;
    int k   = r2 & 31;
    const float* ws = (w == 0) ? Wq : (w == 1) ? Wk : Wv;
    WhT[dst] = (_Float16)ws[(size_t)(kc * 32 + k) * NH + n];
  }
}

// ---------------------------------------------------------------------------
// Kernel 1: fused QKV projection with async-DMA'd weight chunks and
// sched-barrier-pinned fragment pipelining.
// ---------------------------------------------------------------------------
struct XR { f4 a, b, c, d; };

__global__ __launch_bounds__(256) void qkv_project(
    const float* __restrict__ x, const _Float16* __restrict__ WhT,
    _Float16* __restrict__ Qh, _Float16* __restrict__ Kh,
    _Float16* __restrict__ Vt)
{
  __shared__ _Float16 lw[2 * 3 * 2048];      // 24 KB, LDS offset 0

  const int tid  = threadIdx.x;
  const int wave = tid >> 5;
  const int lane = tid & 31;
  const int ln   = lane & 15;
  const int hg   = lane >> 4;
  const int rowBase = blockIdx.x * 128 + wave * 16;

  // LDS is written only by async-DMA (invisible to the optimizer): plant a
  // never-taken, opaquely-guarded store so loads can't fold to undef.
  if (blockIdx.x == 0xFFFFFFFFu) lw[tid] = (_Float16)1.0f;

  v8f zero = {};
  v8f acc[12];
#pragma unroll
  for (int i = 0; i < 12; ++i) acc[i] = zero;

  auto issue_w = [&](int kcIdx) {
    int buf = kcIdx & 1;
#pragma unroll
    for (int w = 0; w < 3; ++w) {
      uint32_t dst = (uint32_t)(buf * 12288 + w * 4096 + tid * 16);
      const char* src =
          (const char*)(WhT + (size_t)(w * 32 + kcIdx) * 2048) + (size_t)tid * 16;
      asm volatile("global_load_async_to_lds_b128 %0, %1, off"
                   :: "v"(dst), "v"(src) : "memory");
    }
  };

  // Raw x loads for one chunk: runs K = hg*8+e and 16+hg*8+e.
  auto loadX = [&](int kcIdx) {
    const float* xr = x + (size_t)(rowBase + ln) * NC + kcIdx * 32 + hg * 8;
    XR r;
    r.a = *(const f4*)(xr);
    r.b = *(const f4*)(xr + 4);
    r.c = *(const f4*)(xr + 16);
    r.d = *(const f4*)(xr + 20);
    return r;
  };
  auto cvtX = [&](const XR& r) {
    v16h a;
#pragma unroll
    for (int e = 0; e < 4; ++e) {
      ((_Float16*)&a)[e]      = (_Float16)r.a[e];
      ((_Float16*)&a)[4 + e]  = (_Float16)r.b[e];
      ((_Float16*)&a)[8 + e]  = (_Float16)r.c[e];
      ((_Float16*)&a)[12 + e] = (_Float16)r.d[e];
    }
    return a;
  };

  issue_w(0);
  XR xcur = loadX(0);

  for (int kcIdx = 0; kcIdx < 32; ++kcIdx) {
    if (kcIdx + 1 < 32) {
      issue_w(kcIdx + 1);                    // prefetch next weight chunk
      asm volatile("s_wait_asynccnt 0x3" ::: "memory");
    } else {
      asm volatile("s_wait_asynccnt 0x0" ::: "memory");
    }
    __syncthreads();                         // chunk visible to all waves

    XR xnext;
    if (kcIdx + 1 < 32) xnext = loadX(kcIdx + 1);   // fly under the WMMAs

    v16h afrag = cvtX(xcur);
    __builtin_amdgcn_sched_barrier(0);       // keep x loads issued up here

    const _Float16* lb = &lw[(kcIdx & 1) * 6144];
    auto loadB = [&](int i) {
      int w = i >> 2, nt = i & 3;
      const _Float16* bp = lb + w * 2048 + (nt * 16 + ln) * 32 + hg * 8;
      v16h bf;
      ((h8*)&bf)[0] = *(const h8*)bp;
      ((h8*)&bf)[1] = *(const h8*)(bp + 16);
      return bf;
    };

    // Two groups of 6 fragments: all 12 DS loads of a group are in flight
    // before its 6 WMMAs consume them (sched barriers pin the order).
#pragma unroll
    for (int grp = 0; grp < 2; ++grp) {
      v16h bf[6];
#pragma unroll
      for (int i = 0; i < 6; ++i) bf[i] = loadB(grp * 6 + i);
      __builtin_amdgcn_sched_barrier(0);
#pragma unroll
      for (int i = 0; i < 6; ++i)
        acc[grp * 6 + i] = __builtin_amdgcn_wmma_f32_16x16x32_f16(
            false, afrag, false, bf[i], (short)0, acc[grp * 6 + i], false, false);
      __builtin_amdgcn_sched_barrier(0);
    }

    xcur = xnext;
    __syncthreads();                         // done with this buffer
  }

  // Epilogue.  C/D layout: VGPR r, lane -> (row = r + hg*8, col = ln).
  const float qscale = 0.03125f;             // C^-0.5 folded into Q
  const int b  = rowBase >> 12;
  const int t0 = (rowBase & (NT - 1)) + hg * 8;
#pragma unroll
  for (int nt = 0; nt < 4; ++nt) {
    int col = nt * 16 + ln;
#pragma unroll
    for (int r = 0; r < 8; ++r) {
      int row = rowBase + r + hg * 8;
      Qh[(size_t)row * NH + col] = (_Float16)(acc[0 * 4 + nt][r] * qscale);
      Kh[(size_t)row * NH + col] = (_Float16)(acc[1 * 4 + nt][r]);
    }
    _Float16* vp = Vt + ((size_t)b * NH + col) * NT + t0;
#pragma unroll
    for (int r = 0; r < 8; r += 2) {
      h8 pk2;
      pk2[0] = (_Float16)acc[2 * 4 + nt][r];
      pk2[1] = (_Float16)acc[2 * 4 + nt][r + 1];
      *(uint32_t*)(vp + r) = *(uint32_t*)&pk2;
    }
  }
}

// ---------------------------------------------------------------------------
// Kernel 2: causal flash attention; async K/V staging, WMMA row-sums,
// sched-barrier-pinned fragment loads.
// LDS (bytes): Kbuf[2]@0/4096, Vbuf[2]@8192/12288, P[8 waves]@16384.
// ---------------------------------------------------------------------------
__global__ __launch_bounds__(256) void flash_attn(
    const _Float16* __restrict__ Qh, const _Float16* __restrict__ Kh,
    const _Float16* __restrict__ Vt, float* __restrict__ out)
{
  __shared__ _Float16 lds[12288];            // 24 KB, LDS offset 0

  const int tid  = threadIdx.x;
  const int wave = tid >> 5;
  const int lane = tid & 31;
  const int ln   = lane & 15;
  const int hg   = lane >> 4;

  if (blockIdx.x == 0xFFFFFFFFu) lds[tid] = (_Float16)1.0f;

  const int g  = blockIdx.x * 8 + wave;      // q-tile id
  const int b  = g >> 8;                     // uniform per block
  const int qi = g & 255;
  const size_t qrow0 = (size_t)b * NT + (size_t)qi * 16;

  const _Float16* Kb = Kh + (size_t)b * NT * NH;
  const _Float16* Vb = Vt + (size_t)b * NH * NT;

  v16h qf[2];
  const _Float16* qp = Qh + (qrow0 + ln) * NH;
#pragma unroll
  for (int f = 0; f < 2; ++f) {
    const _Float16* p = qp + f * 32 + hg * 8;
    ((h8*)&qf[f])[0] = *(const h8*)p;
    ((h8*)&qf[f])[1] = *(const h8*)(p + 16);
  }

  v16h ones;
#pragma unroll
  for (int e = 0; e < 16; ++e) ((_Float16*)&ones)[e] = (_Float16)1.0f;

  float m[8];
  v8f zero = {};
  v8f o[4], la;
#pragma unroll
  for (int r = 0; r < 8; ++r) m[r] = -3.0e38f;
#pragma unroll
  for (int i = 0; i < 4; ++i) o[i] = zero;
  la = zero;

  _Float16* Pl = &lds[8192 + wave * 512];

  const int myNc  = ((qi * 16 + 15) + 32) >> 5;
  const int qiMax = ((blockIdx.x & 31) * 8) + 7;
  const int ncMax = ((qiMax * 16 + 15) + 32) >> 5;       // uniform per block

  const int vh  = tid >> 2;
  const int vsg = tid & 3;

  auto issue_chunk = [&](int c) {
    int buf = c & 1;
    {  // K chunk: 32 rows x 64 h = 4KB contiguous.
      uint32_t dst = (uint32_t)(buf * 4096 + tid * 16);
      const char* src = (const char*)Kb + (size_t)c * 4096 + (size_t)tid * 16;
      asm volatile("global_load_async_to_lds_b128 %0, %1, off"
                   :: "v"(dst), "v"(src) : "memory");
    }
    {  // V chunk: 64 h-rows x 32 k from V^T.
      uint32_t dst = (uint32_t)(8192 + buf * 4096 + vh * 64 + vsg * 16);
      const char* src =
          (const char*)Vb + ((size_t)vh * NT + (size_t)c * 32) * 2 + vsg * 16;
      asm volatile("global_load_async_to_lds_b128 %0, %1, off"
                   :: "v"(dst), "v"(src) : "memory");
    }
  };

  issue_chunk(0);

  for (int c = 0; c < ncMax; ++c) {
    if (c + 1 < ncMax) {
      issue_chunk(c + 1);
      asm volatile("s_wait_asynccnt 0x2" ::: "memory");
    } else {
      asm volatile("s_wait_asynccnt 0x0" ::: "memory");
    }
    __syncthreads();

    if (c < myNc) {
      const int k0 = c * 32;
      const _Float16* Kl = &lds[(c & 1) * 2048];          // [krow][h]
      const _Float16* Vl = &lds[4096 + (c & 1) * 2048];   // [h][k 0..31]

      // All 4 K fragments (kt x f) in flight before the S WMMAs.
      v16h kf[4];
#pragma unroll
      for (int i = 0; i < 4; ++i) {
        int kt = i >> 1, f = i & 1;
        const _Float16* p = Kl + (kt * 16 + ln) * 64 + f * 32 + hg * 8;
        ((h8*)&kf[i])[0] = *(const h8*)p;
        ((h8*)&kf[i])[1] = *(const h8*)(p + 16);
      }
      __builtin_amdgcn_sched_barrier(0);

      v8f s[2];
#pragma unroll
      for (int kt = 0; kt < 2; ++kt) {
        v8f a = zero;
#pragma unroll
        for (int f = 0; f < 2; ++f)
          a = __builtin_amdgcn_wmma_f32_16x16x32_f16(
              false, qf[f], false, kf[kt * 2 + f], (short)0, a, false, false);
        s[kt] = a;
      }

      if (k0 + 31 > qi * 16) {               // causal mask near diagonal
#pragma unroll
        for (int kt = 0; kt < 2; ++kt) {
          int kglob = k0 + kt * 16 + ln;
#pragma unroll
          for (int r = 0; r < 8; ++r) {
            int qglob = qi * 16 + r + hg * 8;
            if (kglob > qglob) s[kt][r] = -3.0e38f;
          }
        }
      }

      // V fragments are independent of softmax: issue their DS loads now so
      // they complete under the softmax VALU work (pinned by sched barrier).
      v16h vf[4];
#pragma unroll
      for (int nt = 0; nt < 4; ++nt) {
        const _Float16* p = Vl + (nt * 16 + ln) * 32 + hg * 8;
        ((h8*)&vf[nt])[0] = *(const h8*)p;
        ((h8*)&vf[nt])[1] = *(const h8*)(p + 16);
      }
      __builtin_amdgcn_sched_barrier(0);

      // Online softmax max; sums come from the WMMA row-sum below.
      float alpha[8];
#pragma unroll
      for (int r = 0; r < 8; ++r) {
        float mx = fmaxf(s[0][r], s[1][r]);
        mx = fmaxf(mx, __shfl_xor(mx, 1, 32));
        mx = fmaxf(mx, __shfl_xor(mx, 2, 32));
        mx = fmaxf(mx, __shfl_xor(mx, 4, 32));
        mx = fmaxf(mx, __shfl_xor(mx, 8, 32));
        float mn = fmaxf(m[r], mx);
        alpha[r] = __expf(m[r] - mn);
        m[r] = mn;
        s[0][r] = __expf(s[0][r] - mn);
        s[1][r] = __expf(s[1][r] - mn);
      }

      // P: C-layout -> LDS [row][col] -> A-layout (same-wave transpose).
#pragma unroll
      for (int kt = 0; kt < 2; ++kt)
#pragma unroll
        for (int r = 0; r < 8; ++r)
          Pl[(r + hg * 8) * 32 + kt * 16 + ln] = (_Float16)s[kt][r];

      asm volatile("s_wait_dscnt 0x0" ::: "memory");

      v16h pf;
      {
        const _Float16* p = Pl + ln * 32 + hg * 8;
        ((h8*)&pf)[0] = *(const h8*)p;
        ((h8*)&pf)[1] = *(const h8*)(p + 16);
      }

      // Rescale running O and l; l += rowsum(P) via P x ones.
#pragma unroll
      for (int r = 0; r < 8; ++r) la[r] *= alpha[r];
      la = __builtin_amdgcn_wmma_f32_16x16x32_f16(
          false, pf, false, ones, (short)0, la, false, false);

#pragma unroll
      for (int nt = 0; nt < 4; ++nt) {
#pragma unroll
        for (int r = 0; r < 8; ++r) o[nt][r] *= alpha[r];
        o[nt] = __builtin_amdgcn_wmma_f32_16x16x32_f16(
            false, pf, false, vf[nt], (short)0, o[nt], false, false);
      }
    }

    __syncthreads();   // all waves done with buf (c&1) before reload
  }

  // Normalize and store f32 out[B,T,H].
  float* op = out + qrow0 * NH;
#pragma unroll
  for (int r = 0; r < 8; ++r) {
    float inv = 1.0f / la[r];
#pragma unroll
    for (int nt = 0; nt < 4; ++nt)
      op[(size_t)(r + hg * 8) * NH + nt * 16 + ln] = o[nt][r] * inv;
  }
}

// ---------------------------------------------------------------------------
extern "C" void kernel_launch(void* const* d_in, const int* in_sizes, int n_in,
                              void* d_out, int out_size, void* d_ws, size_t ws_size,
                              hipStream_t stream) {
  (void)in_sizes; (void)n_in; (void)out_size; (void)ws_size;
  const float* x  = (const float*)d_in[0];
  const float* Wk = (const float*)d_in[1];
  const float* Wq = (const float*)d_in[2];
  const float* Wv = (const float*)d_in[3];
  float* out = (float*)d_out;

  _Float16* Qh  = (_Float16*)d_ws;                      // 2 MB
  _Float16* Kh  = Qh + (size_t)NB * NT * NH;            // 2 MB
  _Float16* Vt  = Kh + (size_t)NB * NT * NH;            // 2 MB
  _Float16* WhT = Vt + (size_t)NB * NH * NT;            // 384 KB

  conv_weights<<<dim3(192), dim3(256), 0, stream>>>(Wq, Wk, Wv, WhT);
  qkv_project<<<dim3((NB * NT) / 128), dim3(256), 0, stream>>>(
      x, WhT, Qh, Kh, Vt);
  flash_attn<<<dim3((NB * (NT / 16)) / 8), dim3(256), 0, stream>>>(
      Qh, Kh, Vt, out);
}
